// RFMDSConv_14431090115258
// MI455X (gfx1250) — matrix operand order repaired and checked by reading, not compiled, not run on hardware
//
#include <hip/hip_runtime.h>
#include <math.h>
#include <stdint.h>

typedef __attribute__((ext_vector_type(16))) _Float16 v16h;
typedef __attribute__((ext_vector_type(8)))  _Float16 v8h;
typedef __attribute__((ext_vector_type(8)))  float    v8f;
typedef __attribute__((ext_vector_type(4)))  unsigned int u32x4;
typedef __attribute__((ext_vector_type(8)))  int      i32x8;
typedef __attribute__((ext_vector_type(4)))  int      i32x4;

#define C_   128
#define H0   64
#define W0   64
#define HW0  4096
#define B_   8
#define H2   192
#define W2   192
#define HW2  36864          // 192*192
#define NE   37748736       // 8*128*192*192
#define O_   256
#define KF   1152           // 128*9

// ---------------- Tensor Data Mover: 2D f16 tile -> LDS ----------------
// Descriptor per CDNA5 ISA ch8: group0 {count=1, lds_addr, global_addr, type=2},
// group1 {data_size=2B, tensor_dim0/1, tile_dim0/1, tensor_dim0_stride}.
__device__ inline void tdm_load_2d_f16(unsigned ldsOff, const _Float16* gptr,
                                       unsigned tdim0, unsigned tdim1,
                                       unsigned tile0, unsigned tile1,
                                       unsigned long long stride0) {
  unsigned long long ga = (unsigned long long)(uintptr_t)gptr;
  u32x4 g0 = { 1u,                                   // count=1, user desc
               ldsOff,                               // lds_addr (bytes)
               (unsigned)ga,                         // global_addr[31:0]
               (unsigned)(ga >> 32) | (2u << 30) };  // global_addr[56:32], type=2
  int w0 = (int)(1u << 16);                          // data_size=1 (2 bytes)
  int w1 = (int)((tdim0 & 0xFFFFu) << 16);           // tensor_dim0[15:0]
  int w2 = (int)((tdim0 >> 16) | ((tdim1 & 0xFFFFu) << 16));
  int w3 = (int)((tdim1 >> 16) | ((tile0 & 0xFFFFu) << 16));
  int w4 = (int)(tile1 & 0xFFFFu);                   // tile_dim1 (tile_dim2=0)
  int w5 = (int)(unsigned)(stride0 & 0xFFFFFFFFu);   // tensor_dim0_stride[31:0]
  int w6 = (int)(unsigned)(stride0 >> 32);           // stride[47:32]
  i32x8 g1 = { w0, w1, w2, w3, w4, w5, w6, 0 };
  i32x4 z4 = { 0, 0, 0, 0 };
#if defined(__clang_major__) && __clang_major__ >= 23
  i32x8 z8 = { 0, 0, 0, 0, 0, 0, 0, 0 };
  __builtin_amdgcn_tensor_load_to_lds(g0, g1, z4, z4, z8, 0);
#else
  __builtin_amdgcn_tensor_load_to_lds(g0, g1, z4, z4, 0);
#endif
}

// ---------------- SE attention ----------------
__global__ void k_gap(const float* __restrict__ x, float* __restrict__ y) {
  __shared__ float red[256];
  int bc = blockIdx.x;
  const float* p = x + (long)bc * HW0;
  float a = 0.f;
  for (int i = threadIdx.x; i < HW0; i += 256) a += p[i];
  red[threadIdx.x] = a; __syncthreads();
  for (int s = 128; s > 0; s >>= 1) {
    if (threadIdx.x < s) red[threadIdx.x] += red[threadIdx.x + s];
    __syncthreads();
  }
  if (threadIdx.x == 0) y[bc] = red[0] * (1.f / HW0);
}

__global__ void k_se(const float* __restrict__ y, const float* __restrict__ w1,
                     const float* __restrict__ w2, float* __restrict__ ca) {
  __shared__ float hid[16];
  int t = threadIdx.x;
  for (int b = 0; b < B_; b++) {
    if (t < 16) {
      float a = 0.f;
      for (int c = 0; c < C_; c++) a += y[b * C_ + c] * w1[t * C_ + c];
      hid[t] = fmaxf(a, 0.f);
    }
    __syncthreads();
    if (t < C_) {
      float a = 0.f;
      for (int r = 0; r < 16; r++) a += hid[r] * w2[t * 16 + r];
      ca[b * C_ + t] = 1.f / (1.f + expf(-a));
    }
    __syncthreads();
  }
}

// ------- one-time weight convert: pw_w -> f16; fc_w -> f16 permuted [o][r][c] -------
__global__ void k_cvtw(const float* __restrict__ pw_w, const float* __restrict__ fc_w,
                       _Float16* __restrict__ pwh, _Float16* __restrict__ fch) {
  int idx = blockIdx.x * 256 + threadIdx.x;        // < 311296
  if (idx < 16384) pwh[idx] = (_Float16)pw_w[idx];
  int j = idx - 16384;
  if (j >= 0 && j < O_ * KF) {
    int o = j / KF, k = j - o * KF;
    int r = k >> 7, c = k & 127;                   // k' = r*128 + c
    fch[j] = (_Float16)fc_w[(o * C_ + c) * 9 + r];
  }
}

// ------- gen conv + BN + ReLU + rearrange + *ca  ->  unfold (f32 [b][c][pix]) -------
__global__ __launch_bounds__(256) void k_gen(
    const float* __restrict__ x, const float* __restrict__ gw,
    const float* __restrict__ gg, const float* __restrict__ gb,
    const float* __restrict__ gm, const float* __restrict__ gv,
    const float* __restrict__ ca, float* __restrict__ unf) {
  long idx = (long)blockIdx.x * 256 + threadIdx.x;
  int X = (int)(idx % W2); long t = idx / W2;
  int Y = (int)(t % H2);   t /= H2;
  int c = (int)(t % C_);   int b = (int)(t / C_);
  int n1 = Y % 3, n2 = X % 3, y = Y / 3, xx = X / 3;
  int cc = c * 9 + n1 * 3 + n2;
  const float* wp = gw + cc * 9;
  const float* xp = x + ((long)(b * C_ + c)) * HW0;
  float acc = 0.f;
  for (int kh = 0; kh < 3; kh++) {
    int yy = y + kh - 1; if (yy < 0 || yy >= H0) continue;
    for (int kw = 0; kw < 3; kw++) {
      int xw = xx + kw - 1; if (xw < 0 || xw >= W0) continue;
      acc += wp[kh * 3 + kw] * xp[yy * W0 + xw];
    }
  }
  float sc = gg[cc] * rsqrtf(gv[cc] + 1e-5f);
  float v  = fmaxf(acc * sc + (gb[cc] - gm[cc] * sc), 0.f);
  unf[idx] = v * ca[b * C_ + c];
}

// ---------------- depthwise 5x5 pad 2 -> xi (f32 [b][c][pix]) ----------------
__global__ __launch_bounds__(256) void k_dw5(
    const float* __restrict__ in, const float* __restrict__ w,
    const float* __restrict__ bias, float* __restrict__ out) {
  long idx = (long)blockIdx.x * 256 + threadIdx.x;
  int X = (int)(idx % W2); long t = idx / W2;
  int Y = (int)(t % H2);
  int c = (int)((t / H2) % C_);
  const float* ch = in + (idx - X - (long)Y * W2);
  const float* wp = w + c * 25;
  float acc = bias[c];
  for (int kh = 0; kh < 5; kh++) {
    int yy = Y + kh - 2; if (yy < 0 || yy >= H2) continue;
    const float* row = ch + (long)yy * W2;
    for (int kw = 0; kw < 5; kw++) {
      int xx = X + kw - 2; if (xx < 0 || xx >= W2) continue;
      acc += wp[kh * 5 + kw] * row[xx];
    }
  }
  out[idx] = acc;
}

// ------- fused separable branch -> s accumulated as f16 [b][pix][c] -------
__global__ __launch_bounds__(256) void k_branch(
    const float* __restrict__ xi, const float* __restrict__ wh,
    const float* __restrict__ bh, const float* __restrict__ wv,
    const float* __restrict__ bv, int ks, int pad,
    _Float16* __restrict__ sbuf, int initFlag) {
  long idx = (long)blockIdx.x * 256 + threadIdx.x;
  int X = (int)(idx % W2); long t = idx / W2;
  int Y = (int)(t % H2);
  int c = (int)((t / H2) % C_);
  int b = (int)(t / ((long)H2 * C_));
  const float* ch = xi + (idx - X - (long)Y * W2);
  const float* whc = wh + c * ks;
  const float* wvc = wv + c * ks;
  float acc = bv[c];
  for (int i = 0; i < ks; i++) {
    int yy = Y + i - pad; if (yy < 0 || yy >= H2) continue;
    const float* row = ch + (long)yy * W2;
    float h = bh[c];
    for (int j = 0; j < ks; j++) {
      int xx = X + j - pad; if (xx < 0 || xx >= W2) continue;
      h += whc[j] * row[xx];
    }
    acc += wvc[i] * h;
  }
  long so = ((long)b * HW2 + (long)Y * W2 + X) * C_ + c;   // [b][pix][c]
  if (initFlag) sbuf[so] = (_Float16)(xi[idx] + acc);
  else          sbuf[so] = (_Float16)((float)sbuf[so] + acc);
}

// ---------------- WMMA fragment helpers ----------------
__device__ inline v16h make_a_frag(const _Float16* row, int half) {
  const _Float16* p = row + half * 16;
  v8h lo = *(const v8h*)(p);
  v8h hi = *(const v8h*)(p + 8);
  return __builtin_shufflevector(lo, hi, 0,1,2,3,4,5,6,7,8,9,10,11,12,13,14,15);
}

// ------- pointwise 1x1 conv GEMM; double-buffered TDM pipeline; epilogue (acc+b)*unfold -------
__global__ __launch_bounds__(256) void k_pw(
    const _Float16* __restrict__ inh,   // [B][HW2][C] f16
    const _Float16* __restrict__ wh,    // [C][C] f16
    const float* __restrict__ bias,     // [C]
    const float* __restrict__ unf,      // [B][C][HW2] f32
    _Float16* __restrict__ outh) {      // [B][HW2][C] f16
  __shared__ _Float16 As[2][C_ * 32];   // 2 x 8KB  [m][k]
  __shared__ _Float16 Bs[2][128 * 32];  // 2 x 8KB  [n][k]
  int pbase = blockIdx.x * 128;
  int b = blockIdx.y;
  int tid = threadIdx.x, l = tid & 31, wv = tid >> 5;
  int half = l >> 4, mr = l & 15;
  if (tid > 10000) { As[0][0] = (_Float16)0; Bs[0][0] = (_Float16)0; }
  unsigned asOff = (unsigned)(uintptr_t)&As[0][0];
  unsigned bsOff = (unsigned)(uintptr_t)&Bs[0][0];
  const unsigned aSz = C_ * 32 * 2, bSz = 128 * 32 * 2;   // bytes per buffer
  long inBase = ((long)b * HW2 + pbase) * C_;
  if (tid < 32) {                       // preload tile 0
    tdm_load_2d_f16(asOff, wh, C_, C_, 32, C_, C_);
    tdm_load_2d_f16(bsOff, inh + inBase, C_, 128, 32, 128, C_);
  }
  v8f acc[8] = {};
  for (int it = 0; it < 4; it++) {
    int cur = it & 1;
    if (tid < 32) {
      if (it < 3) {                     // issue next tile into alt buffer
        int kn = (it + 1) * 32;
        tdm_load_2d_f16(asOff + (cur ^ 1) * aSz, wh + kn, C_, C_, 32, C_, C_);
        tdm_load_2d_f16(bsOff + (cur ^ 1) * bSz, inh + inBase + kn, C_, 128, 32, 128, C_);
        __builtin_amdgcn_s_wait_tensorcnt((short)2);  // current pair done, next in flight
      } else {
        __builtin_amdgcn_s_wait_tensorcnt((short)0);
      }
    }
    __syncthreads();
    const _Float16* A  = &As[cur][0];
    const _Float16* Bp = &Bs[cur][0];
    v16h af = make_a_frag(&A[(wv * 16 + mr) * 32], half);
    for (int nt = 0; nt < 8; nt++) {
      v16h bf = *(const v16h*)&Bp[(nt * 16 + mr) * 32 + half * 16];
      acc[nt] = __builtin_amdgcn_wmma_f32_16x16x32_f16(
          false, af, false, bf, (short)0, acc[nt], false, false);
    }
    __syncthreads();                    // retire readers before buffer reuse
  }
  long bOffC = (long)b * C_ * HW2;
  int m0 = wv * 16 + half * 8;
  for (int nt = 0; nt < 8; nt++) {
    int n = pbase + nt * 16 + mr;
    v8h ph;
    for (int j = 0; j < 8; j++) {
      float v = (acc[nt][j] + bias[m0 + j]) * unf[bOffC + (long)(m0 + j) * HW2 + n];
      ph[j] = (_Float16)v;
    }
    *(v8h*)&outh[((long)b * HW2 + n) * C_ + m0] = ph;
  }
}

// ------- final conv: 3x3 stride 3 GEMM (K reordered r*128+c); pipelined TDM; BN+ReLU -------
__global__ __launch_bounds__(256) void k_final(
    const _Float16* __restrict__ cd,    // conv_data [B][HW2][C] f16
    const _Float16* __restrict__ fwh,   // [256][1152] f16, k' = r*128+c
    const float* __restrict__ fb,
    const float* __restrict__ bng, const float* __restrict__ bnb,
    const float* __restrict__ bnm, const float* __restrict__ bnv,
    float* __restrict__ out) {          // [B][256][64][64]
  __shared__ _Float16 As[2][O_ * 32];   // 2 x 16KB
  __shared__ _Float16 Bs[2][64 * 32];   // 2 x 4KB
  int y = blockIdx.x;                   // output row; tile = 64 pixels (x=0..63)
  int pbase = y * 64;
  int b = blockIdx.y;
  int tid = threadIdx.x, l = tid & 31, wv = tid >> 5;
  int half = l >> 4, mr = l & 15;
  if (tid > 10000) { As[0][0] = (_Float16)0; Bs[0][0] = (_Float16)0; }
  unsigned asOff = (unsigned)(uintptr_t)&As[0][0];
  unsigned bsOff = (unsigned)(uintptr_t)&Bs[0][0];
  const unsigned aSz = O_ * 32 * 2, bSz = 64 * 32 * 2;
  long cdB = (long)b * HW2 * C_;
  const int NK = KF / 32;               // 36
  auto issue = [&](int k0, int sel) {
    int r = k0 >> 7, c0 = k0 & 127;     // fixed (kh,kw) within a K step
    int kh = r / 3, kw = r - kh * 3;
    tdm_load_2d_f16(asOff + sel * aSz, fwh + k0, KF, O_, 32, O_, KF);
    tdm_load_2d_f16(bsOff + sel * bSz,
                    cd + cdB + ((long)(3 * y + kh) * W2 + kw) * C_ + c0,
                    3 * C_, 64, 32, 64, 3 * C_);
  };
  if (tid < 32) issue(0, 0);            // preload tile 0
  v8f acc[8] = {};                      // 2 m-tiles x 4 n-tiles
  for (int it = 0; it < NK; it++) {
    int cur = it & 1;
    if (tid < 32) {
      if (it + 1 < NK) {
        issue((it + 1) * 32, cur ^ 1);
        __builtin_amdgcn_s_wait_tensorcnt((short)2);
      } else {
        __builtin_amdgcn_s_wait_tensorcnt((short)0);
      }
    }
    __syncthreads();
    const _Float16* A  = &As[cur][0];
    const _Float16* Bp = &Bs[cur][0];
    for (int mt = 0; mt < 2; mt++) {
      v16h af = make_a_frag(&A[((wv * 2 + mt) * 16 + mr) * 32], half);
      for (int nt = 0; nt < 4; nt++) {
        v16h bf = *(const v16h*)&Bp[(nt * 16 + mr) * 32 + half * 16];
        acc[mt * 4 + nt] = __builtin_amdgcn_wmma_f32_16x16x32_f16(
            false, af, false, bf, (short)0, acc[mt * 4 + nt], false, false);
      }
    }
    __syncthreads();
  }
  long oB = (long)b * O_ * HW0;
  for (int mt = 0; mt < 2; mt++) {
    for (int j = 0; j < 8; j++) {
      int o = (wv * 2 + mt) * 16 + half * 8 + j;
      float sc = bng[o] * rsqrtf(bnv[o] + 1e-5f);
      float sh = bnb[o] - bnm[o] * sc;
      float bi = fb[o];
      for (int nt = 0; nt < 4; nt++) {
        int n = pbase + nt * 16 + mr;
        float v = (acc[mt * 4 + nt][j] + bi) * sc + sh;
        out[oB + (long)o * HW0 + n] = fmaxf(v, 0.f);
      }
    }
  }
}

extern "C" void kernel_launch(void* const* d_in, const int* in_sizes, int n_in,
                              void* d_out, int out_size, void* d_ws, size_t ws_size,
                              hipStream_t stream) {
  const float* x      = (const float*)d_in[0];
  const float* se_w1  = (const float*)d_in[1];
  const float* se_w2  = (const float*)d_in[2];
  const float* gen_w  = (const float*)d_in[3];
  const float* gen_g  = (const float*)d_in[4];
  const float* gen_b  = (const float*)d_in[5];
  const float* gen_m  = (const float*)d_in[6];
  const float* gen_v  = (const float*)d_in[7];
  const float* w5     = (const float*)d_in[8];
  const float* b5     = (const float*)d_in[9];
  const float* w17    = (const float*)d_in[10];
  const float* b17    = (const float*)d_in[11];
  const float* w71    = (const float*)d_in[12];
  const float* b71    = (const float*)d_in[13];
  const float* w1x11  = (const float*)d_in[14];
  const float* b1x11  = (const float*)d_in[15];
  const float* w11x1  = (const float*)d_in[16];
  const float* b11x1  = (const float*)d_in[17];
  const float* w1x21  = (const float*)d_in[18];
  const float* b1x21  = (const float*)d_in[19];
  const float* w21x1  = (const float*)d_in[20];
  const float* b21x1  = (const float*)d_in[21];
  const float* pw_w   = (const float*)d_in[22];
  const float* pw_b   = (const float*)d_in[23];
  const float* fc_w   = (const float*)d_in[24];
  const float* fc_b   = (const float*)d_in[25];
  const float* fbn_g  = (const float*)d_in[26];
  const float* fbn_b  = (const float*)d_in[27];
  const float* fbn_m  = (const float*)d_in[28];
  const float* fbn_v  = (const float*)d_in[29];

  char* base = (char*)d_ws;
  float*    buf0  = (float*)base;                          // unfold f32   (NE*4)
  float*    buf1  = (float*)(base + (size_t)NE * 4);       // xi f32, later t f16
  _Float16* buf1h = (_Float16*)buf1;
  _Float16* buf2h = (_Float16*)(base + (size_t)NE * 8);    // s f16, later conv_data f16
  float*    yv    = (float*)(base + (size_t)NE * 10);      // [1024]
  float*    ca    = yv + 1024;                             // [1024]
  _Float16* pwh   = (_Float16*)(ca + 1024);                // [16384]
  _Float16* fch   = pwh + 16384;                           // [294912]

  const int EB = NE / 256;

  k_cvtw<<<dim3(1216), dim3(256), 0, stream>>>(pw_w, fc_w, pwh, fch);
  k_gap <<<dim3(B_ * C_), dim3(256), 0, stream>>>(x, yv);
  k_se  <<<dim3(1), dim3(128), 0, stream>>>(yv, se_w1, se_w2, ca);
  k_gen <<<dim3(EB), dim3(256), 0, stream>>>(x, gen_w, gen_g, gen_b, gen_m, gen_v, ca, buf0);
  k_dw5 <<<dim3(EB), dim3(256), 0, stream>>>(buf0, w5, b5, buf1);

  // s = xi + branch7 + branch11 + branch21   (f16, [b][pix][c], into buf2h)
  k_branch<<<dim3(EB), dim3(256), 0, stream>>>(buf1, w17,   b17,   w71,   b71,   7, 3,  buf2h, 1);
  k_branch<<<dim3(EB), dim3(256), 0, stream>>>(buf1, w1x11, b1x11, w11x1, b11x1, 11, 5, buf2h, 0);
  k_branch<<<dim3(EB), dim3(256), 0, stream>>>(buf1, w1x21, b1x21, w21x1, b21x1, 21, 10, buf2h, 0);

  // t = (pw(s)+b) * unfold          : buf2h -> buf1h
  k_pw<<<dim3(HW2 / 128, B_), dim3(256), 0, stream>>>(buf2h, pwh, pw_b, buf0, buf1h);
  // conv_data = (pw(t)+b) * unfold  : buf1h -> buf2h
  k_pw<<<dim3(HW2 / 128, B_), dim3(256), 0, stream>>>(buf1h, pwh, pw_b, buf0, buf2h);

  // final 3x3 stride-3 conv + BN + ReLU
  k_final<<<dim3(HW0 / 64, B_), dim3(256), 0, stream>>>(buf2h, fch, fc_b,
                                                        fbn_g, fbn_b, fbn_m, fbn_v,
                                                        (float*)d_out);
}
